// LGABlock_82910048682081
// MI455X (gfx1250) — matrix operand order repaired and checked by reading, not compiled, not run on hardware
//
#include <hip/hip_runtime.h>
#include <hip/hip_bf16.h>
#include <math.h>

// ---------------------------------------------------------------------------
// LGA block for MI455X (gfx1250): all dense GEMMs on v_wmma_f32_16x16x32_f16,
// fragment-order LDS staging (one contiguous v16h LDS load per WMMA operand),
// elementwise stages as bandwidth-bound wave32 kernels.
// ---------------------------------------------------------------------------

typedef __attribute__((ext_vector_type(16))) _Float16 v16h;
typedef __attribute__((ext_vector_type(8)))  float    v8f;

#define Bb 32
#define Nn_TOK 3136
#define Hs 56
#define Ws 56
#define DIM 384
#define HEADS 8
#define HEAD_DIM 48
#define HID 1536
#define MTOK (Bb * Nn_TOK)   // 100352, multiple of 64

// ---------------------------------------------------------------------------
// Weight f32 -> f16 conversion
// ---------------------------------------------------------------------------
__global__ __launch_bounds__(256) void cvt_f32_to_f16(const float* __restrict__ src,
                                                      _Float16* __restrict__ dst,
                                                      size_t n) {
    size_t i = (size_t)blockIdx.x * blockDim.x + threadIdx.x;
    if (i < n) dst[i] = (_Float16)src[i];
}

// ---------------------------------------------------------------------------
// CPE: depthwise 3x3 conv (SAME, groups=C) residual.  x:[B,N,C] -> fA:[B,N,C]
// ---------------------------------------------------------------------------
__global__ __launch_bounds__(256) void cpe_kernel(const float* __restrict__ x,
                                                  const float* __restrict__ w,   // [C,1,3,3]
                                                  const float* __restrict__ bias,
                                                  float* __restrict__ out) {
    size_t idx = (size_t)blockIdx.x * 256 + threadIdx.x;
    if (idx >= (size_t)MTOK * DIM) return;
    int c = (int)(idx % DIM);
    size_t bn = idx / DIM;
    int n = (int)(bn % Nn_TOK);
    size_t bi = bn / Nn_TOK;
    int h = n / Ws, wcol = n % Ws;
    float acc = bias[c];
    #pragma unroll
    for (int dh = -1; dh <= 1; ++dh) {
        int hh = h + dh;
        if (hh < 0 || hh >= Hs) continue;
        #pragma unroll
        for (int dw = -1; dw <= 1; ++dw) {
            int ww = wcol + dw;
            if (ww < 0 || ww >= Ws) continue;
            acc += x[((bi * Nn_TOK) + (size_t)(hh * Ws + ww)) * DIM + c]
                 * w[c * 9 + (dh + 1) * 3 + (dw + 1)];
        }
    }
    out[idx] = x[idx] + acc;
}

// ---------------------------------------------------------------------------
// LayerNorm over DIM=384; one wave32 per token (12 elems/lane), output f16.
// ---------------------------------------------------------------------------
__global__ __launch_bounds__(128) void ln_kernel(const float* __restrict__ x,
                                                 const float* __restrict__ w,
                                                 const float* __restrict__ b,
                                                 _Float16* __restrict__ out) {
    int wave = threadIdx.x >> 5, lane = threadIdx.x & 31;
    int token = blockIdx.x * 4 + wave;
    if (token >= MTOK) return;
    const float* row = x + (size_t)token * DIM;
    float vals[12];
    float s = 0.f;
    #pragma unroll
    for (int i = 0; i < 12; ++i) { vals[i] = row[lane + i * 32]; s += vals[i]; }
    #pragma unroll
    for (int off = 16; off > 0; off >>= 1) s += __shfl_xor(s, off, 32);
    float mu = s * (1.f / DIM);
    float v = 0.f;
    #pragma unroll
    for (int i = 0; i < 12; ++i) { float d = vals[i] - mu; v += d * d; }
    #pragma unroll
    for (int off = 16; off > 0; off >>= 1) v += __shfl_xor(v, off, 32);
    float r = rsqrtf(v * (1.f / DIM) + 1e-5f);
    _Float16* orow = out + (size_t)token * DIM;
    #pragma unroll
    for (int i = 0; i < 12; ++i) {
        int cc = lane + i * 32;
        orow[cc] = (_Float16)((vals[i] - mu) * r * w[cc] + b[cc]);
    }
}

// ---------------------------------------------------------------------------
// Tiled WMMA GEMM:  C[M,Nn] = A[M,K] * W[Nn,K]^T + bias, fused epilogue.
//   EPI 0: qkv  -> f16 out, relu on cols < 2*DIM (q and k)
//   EPI 1: proj/fc2 -> f32 out = resid + val
//   EPI 2: fc1  -> f16 out = gelu_exact(val)
// Block = 128 threads = 4 waves; block tile 64x64; K tile 64 (2 WMMA k-steps
// per barrier, 8 WMMAs/wave per stage); wave tile 32x32 (2x2 fragments).
//
// LDS tiles are stored in FRAGMENT ORDER [kk][subtile][lane][16 halves] so
// each WMMA operand is a single contiguous 32B v16h LDS load (ds_load_b128
// pair straight into the source VGPRs -- no v_mov_b16 re-packing).
//
// A-fragment lane map (ISA 7.12.2, 16-bit A 16x32):
//   lane = m + 16*((klocal>>3)&1), elem = (klocal>>4)*8 + (klocal&7)
// B-fragment lane map (16-bit B 32x16):
//   lane = n + 16*((klocal>>3)&1), elem = (klocal>>4)*8 + (klocal&7)
// C layout: col = lane&15, row = vgpr + 8*(lane>=16).
// ---------------------------------------------------------------------------
template <int EPI>
__global__ __launch_bounds__(128) void gemm_wmma(const _Float16* __restrict__ A,
                                                 const _Float16* __restrict__ W,
                                                 const float* __restrict__ bias,
                                                 const float* __restrict__ resid,
                                                 float* __restrict__ outF,
                                                 _Float16* __restrict__ outH,
                                                 int M, int K, int Nc) {
    __shared__ alignas(32) _Float16 As[2][4][32][16];   // [kk][s][lane][elem] 8KB
    __shared__ alignas(32) _Float16 Bs[2][4][32][16];   // 8KB

    const int tid  = threadIdx.x;
    const int wave = tid >> 5, lane = tid & 31;
    const int wr = wave >> 1, wc = wave & 1;
    const int rowBase = blockIdx.y * 64;
    const int colBase = blockIdx.x * 64;

    // Staging: thread t covers row lr, halves [(t&1)*32, (t&1)*32+32) of the
    // 64-deep K tile as 4x float4 (8 halves each).
    const int lr  = tid >> 1;
    const int kkS = tid & 1;            // which 32-half (= kk slot)
    const int sS  = lr >> 4;            // subtile 0..3
    const int mS  = lr & 15;            // row/col within subtile

    v8f c[2][2] = {};

    for (int k0 = 0; k0 < K; k0 += 64) {
        const _Float16* ga = A + (size_t)(rowBase + lr) * K + k0 + kkS * 32;
        const _Float16* gw = W + (size_t)(colBase + lr) * K + k0 + kkS * 32;
        float4 av[4], wv[4];
        #pragma unroll
        for (int q = 0; q < 4; ++q) {
            av[q] = *(const float4*)(ga + q * 8);
            wv[q] = *(const float4*)(gw + q * 8);
        }
        if (k0 + 64 < K) {
            __builtin_prefetch(ga + 64, 0, 1);
            __builtin_prefetch(gw + 64, 0, 1);
        }
        __syncthreads();
        // A: 8-half group at klocal base kb in {0,8,16,24}:
        //   lane = mS + 16*((kb>>3)&1), elem base = (kb>>4)*8
        #pragma unroll
        for (int q = 0; q < 4; ++q) {
            const int kb = q * 8;
            *(float4*)&As[kkS][sS][mS + 16 * ((kb >> 3) & 1)][(kb >> 4) * 8] = av[q];
            // B: lane = nS + 16*((kb>>3)&1), elem base = (kb>>4)*8
            *(float4*)&Bs[kkS][sS][mS + 16 * ((kb >> 3) & 1)][(kb >> 4) * 8] = wv[q];
        }
        __syncthreads();

        #pragma unroll
        for (int kk = 0; kk < 2; ++kk) {
            v16h bf[2];
            #pragma unroll
            for (int j = 0; j < 2; ++j)
                bf[j] = *(const v16h*)&Bs[kk][wc * 2 + j][lane][0];
            #pragma unroll
            for (int i = 0; i < 2; ++i) {
                v16h a = *(const v16h*)&As[kk][wr * 2 + i][lane][0];
                #pragma unroll
                for (int j = 0; j < 2; ++j) {
                    c[i][j] = __builtin_amdgcn_wmma_f32_16x16x32_f16(
                        false, a, false, bf[j], (short)0, c[i][j], false, false);
                }
            }
        }
    }

    // Epilogue.
    const int mlan = lane & 15;
    const int lrow = (lane >> 4) * 8;
    #pragma unroll
    for (int i = 0; i < 2; ++i) {
        #pragma unroll
        for (int j = 0; j < 2; ++j) {
            #pragma unroll
            for (int v = 0; v < 8; ++v) {
                int row = rowBase + wr * 32 + i * 16 + lrow + v;
                int col = colBase + wc * 32 + j * 16 + mlan;
                float val = c[i][j][v] + bias[col];
                size_t idx = (size_t)row * Nc + col;
                if constexpr (EPI == 0) {
                    outH[idx] = (_Float16)(col < 2 * DIM ? fmaxf(val, 0.f) : val);
                } else if constexpr (EPI == 1) {
                    outF[idx] = resid[idx] + val;
                } else {
                    outH[idx] = (_Float16)(0.5f * val * (1.f + erff(val * 0.70710678f)));
                }
            }
        }
    }
}

// ---------------------------------------------------------------------------
// k_sum[b,h,d] = sum_n relu_k[b,n,h,d]   (k = qkv cols [384,768))
// ---------------------------------------------------------------------------
__global__ __launch_bounds__(64) void ksum_kernel(const _Float16* __restrict__ qkv,
                                                  float* __restrict__ ksum) {
    int bh = blockIdx.x;
    int b = bh / HEADS, hh = bh % HEADS;
    int d = threadIdx.x;
    if (d >= HEAD_DIM) return;
    const _Float16* base = qkv + (size_t)b * Nn_TOK * (3 * DIM) + DIM + hh * HEAD_DIM + d;
    float s = 0.f;
    for (int n = 0; n < Nn_TOK; ++n) s += (float)base[(size_t)n * (3 * DIM)];
    ksum[bh * HEAD_DIM + d] = s;
}

// ---------------------------------------------------------------------------
// kv[b,h,d,e] = temperature[h] * sum_n k[b,h,n,d] * v[b,h,n,e]
// Per (b,h): 48x48 GEMM over K=3136 via WMMA, 4 waves split K,
// cross-wave reduce with LDS ds_add_f32 atomics.
// ---------------------------------------------------------------------------
__global__ __launch_bounds__(128) void kv_kernel(const _Float16* __restrict__ qkv,
                                                 const float* __restrict__ temp,
                                                 float* __restrict__ kv) {
    __shared__ float acc[HEAD_DIM * HEAD_DIM];
    const int bh = blockIdx.x;
    const int b = bh / HEADS, hh = bh % HEADS;
    const int tid = threadIdx.x, wave = tid >> 5, lane = tid & 31;
    const int mlan = lane & 15, g = lane >> 4;

    for (int i = tid; i < HEAD_DIM * HEAD_DIM; i += 128) acc[i] = 0.f;
    __syncthreads();

    const _Float16* kbase = qkv + (size_t)b * Nn_TOK * (3 * DIM) + DIM + hh * HEAD_DIM;
    const _Float16* vbase = qkv + (size_t)b * Nn_TOK * (3 * DIM) + 2 * DIM + hh * HEAD_DIM;

    v8f c[3][3] = {};
    for (int n0 = wave * 32; n0 < Nn_TOK; n0 += 128) {
        v16h bf[3];
        #pragma unroll
        for (int te = 0; te < 3; ++te) {
            #pragma unroll
            for (int q = 0; q < 16; ++q) {
                int n = n0 + g * 16 + q;
                bf[te][q] = vbase[(size_t)n * (3 * DIM) + te * 16 + mlan];
            }
        }
        #pragma unroll
        for (int ti = 0; ti < 3; ++ti) {
            v16h a;   // A = k^T : row d = ti*16+mlan, K-dim = token index
            #pragma unroll
            for (int q = 0; q < 16; ++q) {
                int n = n0 + g * 16 + q;
                a[q] = kbase[(size_t)n * (3 * DIM) + ti * 16 + mlan];
            }
            #pragma unroll
            for (int te = 0; te < 3; ++te) {
                c[ti][te] = __builtin_amdgcn_wmma_f32_16x16x32_f16(
                    false, a, false, bf[te], (short)0, c[ti][te], false, false);
            }
        }
    }

    #pragma unroll
    for (int ti = 0; ti < 3; ++ti)
        #pragma unroll
        for (int te = 0; te < 3; ++te)
            #pragma unroll
            for (int v = 0; v < 8; ++v) {
                int row = ti * 16 + g * 8 + v;
                int col = te * 16 + mlan;
                atomicAdd(&acc[row * HEAD_DIM + col], c[ti][te][v]);
            }
    __syncthreads();

    float t = temp[hh];
    for (int i = tid; i < HEAD_DIM * HEAD_DIM; i += 128)
        kv[(size_t)bh * HEAD_DIM * HEAD_DIM + i] = acc[i] * t;
}

// ---------------------------------------------------------------------------
// attn_out[b,n,h*48+e] = (sum_d q[d]*kv[d,e]) / max(sum_d q[d]*ksum[d], 100)
// ---------------------------------------------------------------------------
__global__ __launch_bounds__(256) void attn_kernel(const _Float16* __restrict__ qkv,
                                                   const float* __restrict__ kv,
                                                   const float* __restrict__ ksum,
                                                   _Float16* __restrict__ out) {
    size_t idx = (size_t)blockIdx.x * 256 + threadIdx.x;
    if (idx >= (size_t)MTOK * DIM) return;
    int ce = (int)(idx % DIM);
    size_t bn = idx / DIM;
    int b = (int)(bn / Nn_TOK);
    int hh = ce / HEAD_DIM, e = ce % HEAD_DIM;
    const _Float16* q  = qkv + bn * (3 * DIM) + hh * HEAD_DIM;   // relu'd already
    const float*    kp = kv   + ((size_t)b * HEADS + hh) * HEAD_DIM * HEAD_DIM;
    const float*    ks = ksum + ((size_t)b * HEADS + hh) * HEAD_DIM;
    float acc = 0.f, den = 0.f;
    #pragma unroll 4
    for (int d = 0; d < HEAD_DIM; ++d) {
        float qd = (float)q[d];
        acc += qd * kp[d * HEAD_DIM + e];
        den += qd * ks[d];
    }
    out[idx] = (_Float16)(acc / fmaxf(den, 100.f));
}

// ---------------------------------------------------------------------------
// Host-side orchestration
// ---------------------------------------------------------------------------
extern "C" void kernel_launch(void* const* d_in, const int* in_sizes, int n_in,
                              void* d_out, int out_size, void* d_ws, size_t ws_size,
                              hipStream_t stream) {
    const float* x      = (const float*)d_in[0];
    const float* cpe_w  = (const float*)d_in[3];
    const float* cpe_b  = (const float*)d_in[4];
    const float* ln1_w  = (const float*)d_in[5];
    const float* ln1_b  = (const float*)d_in[6];
    const float* qkv_w  = (const float*)d_in[7];
    const float* qkv_b  = (const float*)d_in[8];
    const float* temp   = (const float*)d_in[9];
    const float* proj_w = (const float*)d_in[10];
    const float* proj_b = (const float*)d_in[11];
    const float* ln2_w  = (const float*)d_in[12];
    const float* ln2_b  = (const float*)d_in[13];
    const float* fc1_w  = (const float*)d_in[14];
    const float* fc1_b  = (const float*)d_in[15];
    const float* fc2_w  = (const float*)d_in[16];
    const float* fc2_b  = (const float*)d_in[17];
    float* out = (float*)d_out;

    char* ws = (char*)d_ws;
    const size_t M = MTOK;
    float*    fA    = (float*)ws;                                    ws += M * DIM * 4;
    _Float16* hA    = (_Float16*)ws;                                 ws += M * DIM * 2;
    _Float16* hBig  = (_Float16*)ws;                                 ws += M * HID * 2;
    _Float16* wQKVh = (_Float16*)ws;                                 ws += (size_t)3 * DIM * DIM * 2;
    _Float16* wPRJh = (_Float16*)ws;                                 ws += (size_t)DIM * DIM * 2;
    _Float16* wFC1h = (_Float16*)ws;                                 ws += (size_t)HID * DIM * 2;
    _Float16* wFC2h = (_Float16*)ws;                                 ws += (size_t)DIM * HID * 2;
    float*    ksumB = (float*)ws;                                    ws += (size_t)Bb * HEADS * HEAD_DIM * 4;
    float*    kvB   = (float*)ws;

    // 1) weights -> f16
    {
        size_t nq = (size_t)3 * DIM * DIM, np = (size_t)DIM * DIM, n1 = (size_t)HID * DIM;
        cvt_f32_to_f16<<<dim3((unsigned)((nq + 255) / 256)), 256, 0, stream>>>(qkv_w,  wQKVh, nq);
        cvt_f32_to_f16<<<dim3((unsigned)((np + 255) / 256)), 256, 0, stream>>>(proj_w, wPRJh, np);
        cvt_f32_to_f16<<<dim3((unsigned)((n1 + 255) / 256)), 256, 0, stream>>>(fc1_w,  wFC1h, n1);
        cvt_f32_to_f16<<<dim3((unsigned)((n1 + 255) / 256)), 256, 0, stream>>>(fc2_w,  wFC2h, n1);
    }

    const unsigned nElemBlocks = (unsigned)((M * DIM + 255) / 256);

    // 2) CPE residual -> fA
    cpe_kernel<<<nElemBlocks, 256, 0, stream>>>(x, cpe_w, cpe_b, fA);

    // 3) LN1 -> hA (f16)
    ln_kernel<<<(unsigned)(M / 4), 128, 0, stream>>>(fA, ln1_w, ln1_b, hA);

    // 4) QKV GEMM (relu on q,k) -> hBig[:, :1152]
    gemm_wmma<0><<<dim3(3 * DIM / 64, (unsigned)(M / 64)), 128, 0, stream>>>(
        hA, wQKVh, qkv_b, nullptr, nullptr, hBig, (int)M, DIM, 3 * DIM);

    // 5) k_sum, kv = temperature * k^T v
    ksum_kernel<<<Bb * HEADS, 64, 0, stream>>>(hBig, ksumB);
    kv_kernel<<<Bb * HEADS, 128, 0, stream>>>(hBig, temp, kvB);

    // 6) attn_out = (q @ kv) / denom -> hA (f16)
    attn_kernel<<<nElemBlocks, 256, 0, stream>>>(hBig, kvB, ksumB, hA);

    // 7) proj GEMM + residual -> fA (f32, in place)
    gemm_wmma<1><<<dim3(DIM / 64, (unsigned)(M / 64)), 128, 0, stream>>>(
        hA, wPRJh, proj_b, fA, fA, nullptr, (int)M, DIM, DIM);

    // 8) LN2 -> hA
    ln_kernel<<<(unsigned)(M / 4), 128, 0, stream>>>(fA, ln2_w, ln2_b, hA);

    // 9) FC1 GEMM + exact GELU -> hBig (f16)
    gemm_wmma<2><<<dim3(HID / 64, (unsigned)(M / 64)), 128, 0, stream>>>(
        hA, wFC1h, fc1_b, nullptr, nullptr, hBig, (int)M, DIM, HID);

    // 10) FC2 GEMM + residual -> d_out (f32)
    gemm_wmma<1><<<dim3(DIM / 64, (unsigned)(M / 64)), 128, 0, stream>>>(
        hBig, wFC2h, fc2_b, fA, out, nullptr, (int)M, HID, DIM);
}
